// DeepSeekMoE_34849364639780
// MI455X (gfx1250) — matrix operand order repaired and checked by reading, compiled, not run on hardware
//
#include <hip/hip_runtime.h>
#include <hip/hip_bf16.h>

// Problem dims (compile-time constants from the reference)
#define HDIM 1536
#define IDIM 4096
#define NE_SH 2
#define NE_RT 6
#define TTOK 2048

// LDS row stride in halves (80B): keeps b128 frag loads 16B-aligned while
// breaking the 64B bank periodicity (conflict-free staging stores/frag loads)
#define LSTR 40

typedef __attribute__((ext_vector_type(16))) __bf16 v16bf;
typedef __attribute__((ext_vector_type(8)))  float  v8f;

union FragBF {
    v16bf bf;
    uint4 q[2];
};

// gfx1250 async memory->LDS path (ASYNCcnt tracked). Signature per compiler
// diagnostics: (int4 addrspace(1)*, int4 addrspace(3)*, imm offset, imm cpol).
#if defined(__has_builtin)
#if __has_builtin(__builtin_amdgcn_global_load_async_to_lds_b128) && \
    __has_builtin(__builtin_amdgcn_s_wait_asynccnt)
#define USE_ASYNC_LDS 1
#endif
#if __has_builtin(__builtin_amdgcn_cvt_pk_bf16_f32)
#define HAVE_PK_BF16 1
#endif
#endif
#ifndef USE_ASYNC_LDS
#define USE_ASYNC_LDS 0
#endif
#ifndef HAVE_PK_BF16
#define HAVE_PK_BF16 0
#endif

typedef int vi4 __attribute__((vector_size(16)));
typedef __attribute__((address_space(1))) vi4* gptr_vi4;
typedef __attribute__((address_space(3))) vi4* sptr_vi4;

__device__ __forceinline__ void copy16B_to_lds(const unsigned short* g, unsigned short* s) {
#if USE_ASYNC_LDS
    __builtin_amdgcn_global_load_async_to_lds_b128(
        (gptr_vi4)(vi4*)g, (sptr_vi4)(vi4*)s, 0, 0);
#else
    *(uint4*)s = *(const uint4*)g;
#endif
}

__device__ __forceinline__ void lds_copy_wait() {
#if USE_ASYNC_LDS
    __builtin_amdgcn_s_wait_asynccnt(0);
#endif
}

__device__ __forceinline__ unsigned short f2bf(float f) {
    // round-to-nearest-even fp32 -> bf16
    unsigned int u = __builtin_bit_cast(unsigned int, f);
    u += 0x7FFFu + ((u >> 16) & 1u);
    return (unsigned short)(u >> 16);
}

__device__ __forceinline__ unsigned int pack_bf2(float lo, float hi) {
#if HAVE_PK_BF16
    auto r = __builtin_amdgcn_cvt_pk_bf16_f32(lo, hi);   // v_cvt_pk_bf16_f32 (RTNE)
    return __builtin_bit_cast(unsigned int, r);
#else
    return (unsigned int)f2bf(lo) | ((unsigned int)f2bf(hi) << 16);
#endif
}

// ---------------------------------------------------------------- utilities
__global__ __launch_bounds__(256) void moe_zero_f32(float* __restrict__ p, int n) {
    int i = blockIdx.x * blockDim.x + threadIdx.x;
    if (i < n) p[i] = 0.0f;
}

// 4 elements per thread: float4 in, 4 packed bf16 (uint2) out
__global__ __launch_bounds__(256) void moe_cvt_bf16(const float* __restrict__ src,
                                                    unsigned int* __restrict__ dst, int n) {
    int i = blockIdx.x * blockDim.x + threadIdx.x;
    if (i * 4 < n) {
        float4 v = ((const float4*)src)[i];
        dst[i * 2 + 0] = pack_bf2(v.x, v.y);
        dst[i * 2 + 1] = pack_bf2(v.z, v.w);
    }
}

// ---------------------------------------------------------------- router
// One wave32 per token: partial dot-products across lanes, xor-shuffle reduce,
// then lane 0 forms top-2 combine weights directly from the top-2 logits.
__global__ __launch_bounds__(256) void moe_router(const float* __restrict__ x,
                                                  const float* __restrict__ Wr,
                                                  const float* __restrict__ rb,
                                                  float* __restrict__ cw) {
    const int lane = threadIdx.x & 31;
    const int wave = threadIdx.x >> 5;
    const int t = blockIdx.x * 8 + wave;
    if (t >= TTOK) return;

    float p[NE_RT];
#pragma unroll
    for (int e = 0; e < NE_RT; ++e) p[e] = 0.0f;

    for (int h = lane; h < HDIM; h += 32) {
        float xv = x[(size_t)t * HDIM + h];
#pragma unroll
        for (int e = 0; e < NE_RT; ++e) p[e] += xv * Wr[h * NE_RT + e];
    }
#pragma unroll
    for (int e = 0; e < NE_RT; ++e) {
#pragma unroll
        for (int off = 16; off > 0; off >>= 1)
            p[e] += __shfl_xor(p[e], off, 32);
        p[e] += rb[e];
    }
    if (lane == 0) {
        int a = 0;
#pragma unroll
        for (int e = 1; e < NE_RT; ++e) if (p[e] > p[a]) a = e;
        int b = (a == 0) ? 1 : 0;
#pragma unroll
        for (int e = 0; e < NE_RT; ++e) if (e != a && p[e] > p[b]) b = e;
        float eb = __expf(p[b] - p[a]);   // ea = 1
        float inv = 1.0f / (1.0f + eb);
#pragma unroll
        for (int e = 0; e < NE_RT; ++e) {
            float wv = (e == a) ? inv : ((e == b) ? eb * inv : 0.0f);
            cw[t * NE_RT + e] = wv;
        }
    }
}

// ---------------------------------------------------------------- staging
// X tile: 128x32 bf16 -> LDS [row][k] (stride LSTR), async b128 copies.
__device__ __forceinline__ void stage_x(int tid, int mBlk, int k0,
                                        const unsigned short* __restrict__ src, int ld,
                                        unsigned short* __restrict__ sx) {
    int row = tid >> 1;
    int col = (tid & 1) << 4;
    const unsigned short* gp = src + (size_t)(mBlk + row) * ld + k0 + col;
    unsigned short* sp = sx + row * LSTR + col;
    copy16B_to_lds(gp, sp);
    copy16B_to_lds(gp + 8, sp + 8);
}

// Weight tile: 32x128 fp32 -> bf16, K-transposed LDS [n][k] (stride LSTR).
// Each thread owns a 4k x 4n sub-block: 4x float4 loads, register transpose
// with packed bf16 conversion, 4x conflict-free ds_store_b64.
__device__ __forceinline__ void stage_w(int tid, int nBlk, int k0,
                                        const float* __restrict__ W, int ld,
                                        unsigned short* __restrict__ sw) {
    const int lane = tid & 31;
    const int wv   = tid >> 5;
    const int kg = (lane >> 2) << 2;               // k base: 0,4,...,28
    const int n4 = ((wv << 2) + (lane & 3)) << 2;  // n base: 0..124 step 4
    const float* base = W + (size_t)(k0 + kg) * ld + nBlk + n4;
    float4 r0 = *(const float4*)(base);
    float4 r1 = *(const float4*)(base + ld);
    float4 r2 = *(const float4*)(base + 2 * (size_t)ld);
    float4 r3 = *(const float4*)(base + 3 * (size_t)ld);
    *(uint2*)(sw + (n4 + 0) * LSTR + kg) = make_uint2(pack_bf2(r0.x, r1.x), pack_bf2(r2.x, r3.x));
    *(uint2*)(sw + (n4 + 1) * LSTR + kg) = make_uint2(pack_bf2(r0.y, r1.y), pack_bf2(r2.y, r3.y));
    *(uint2*)(sw + (n4 + 2) * LSTR + kg) = make_uint2(pack_bf2(r0.z, r1.z), pack_bf2(r2.z, r3.z));
    *(uint2*)(sw + (n4 + 3) * LSTR + kg) = make_uint2(pack_bf2(r0.w, r1.w), pack_bf2(r2.w, r3.w));
}

// ---------------------------------------------------------------- GEMM 1
// h[T,I](bf16) = silu(x @ Wg) * (x @ Wu); 128x128 tile/block; K-step 32;
// double-buffered LDS, one barrier per K-step, bf16 WMMA 16x16x32.
__global__ __launch_bounds__(256) void moe_gateup(const unsigned short* __restrict__ xbf, // [T,H]
                                                  const float* __restrict__ Wg,           // [H,I]
                                                  const float* __restrict__ Wu,           // [H,I]
                                                  unsigned short* __restrict__ hbuf) {    // [T,I]
    __shared__ unsigned short sX[2][128 * LSTR];
    __shared__ unsigned short sG[2][128 * LSTR];
    __shared__ unsigned short sU[2][128 * LSTR];

    const int tid  = threadIdx.x;
    const int lane = tid & 31;
    const int wave = tid >> 5;
    const int wm = wave & 3;   // 4 waves over M (32 rows each)
    const int wn = wave >> 2;  // 2 waves over N (64 cols each)
    const int mBlk = blockIdx.y * 128;
    const int nBlk = blockIdx.x * 128;
    const int NK = HDIM / 32;

    const v8f vz = {0.f, 0.f, 0.f, 0.f, 0.f, 0.f, 0.f, 0.f};
    v8f accG[2][4], accU[2][4];
#pragma unroll
    for (int m = 0; m < 2; ++m)
#pragma unroll
        for (int n = 0; n < 4; ++n) { accG[m][n] = vz; accU[m][n] = vz; }

    // prologue: stage tile 0
    stage_x(tid, mBlk, 0, xbf, HDIM, sX[0]);
    stage_w(tid, nBlk, 0, Wg, IDIM, sG[0]);
    stage_w(tid, nBlk, 0, Wu, IDIM, sU[0]);

    for (int kt = 0; kt < NK; ++kt) {
        lds_copy_wait();        // my async LDS writes for buf (kt&1) landed
        __syncthreads();        // everyone's staging for buf (kt&1) visible
        const int b = kt & 1;

        if (kt + 1 < NK) {      // overlap: stage next tile into other buffer
            int k1 = (kt + 1) * 32;
            stage_x(tid, mBlk, k1, xbf, HDIM, sX[b ^ 1]);
            stage_w(tid, nBlk, k1, Wg, IDIM, sG[b ^ 1]);
            stage_w(tid, nBlk, k1, Wu, IDIM, sU[b ^ 1]);
            if (kt + 2 < NK) {  // pull K-tile kt+2 toward L2 (global_prefetch_b8)
                __builtin_prefetch(&Wg[(size_t)(kt + 2) * 32 * IDIM + nBlk + tid], 0, 1);
                __builtin_prefetch(&Wu[(size_t)(kt + 2) * 32 * IDIM + nBlk + tid], 0, 1);
            }
        }

        // A fragments: lane<16 -> K0-15; lane>=16 -> K16-31
        const unsigned short* sx = sX[b];
        FragBF a[2];
#pragma unroll
        for (int m = 0; m < 2; ++m) {
            int row = wm * 32 + m * 16 + (lane & 15);
            int kh = (lane >> 4) << 4;
            a[m].q[0] = *(const uint4*)(sx + row * LSTR + kh);
            a[m].q[1] = *(const uint4*)(sx + row * LSTR + kh + 8);
        }
        const unsigned short* sg = sG[b];
        const unsigned short* su = sU[b];
#pragma unroll
        for (int n = 0; n < 4; ++n) {
            int colb = wn * 64 + n * 16 + (lane & 15);
            int kh = (lane >> 4) << 4;  // B: lane<16 -> K0-15, lane>=16 -> K16-31
            FragBF bg, bu;
            const uint4* pg = (const uint4*)(sg + colb * LSTR + kh);
            bg.q[0] = pg[0]; bg.q[1] = pg[1];
            const uint4* pu = (const uint4*)(su + colb * LSTR + kh);
            bu.q[0] = pu[0]; bu.q[1] = pu[1];
#pragma unroll
            for (int m = 0; m < 2; ++m) {
                accG[m][n] = __builtin_amdgcn_wmma_f32_16x16x32_bf16(
                    false, a[m].bf, false, bg.bf, (short)0, accG[m][n], false, false);
                accU[m][n] = __builtin_amdgcn_wmma_f32_16x16x32_bf16(
                    false, a[m].bf, false, bu.bf, (short)0, accU[m][n], false, false);
            }
        }
    }

    // epilogue: h = silu(g)*u  (C/D layout: M = v + 8*(lane>=16), N = lane&15)
    const int mo = (lane >> 4) << 3;
    const int nn = lane & 15;
#pragma unroll
    for (int m = 0; m < 2; ++m)
#pragma unroll
        for (int n = 0; n < 4; ++n)
#pragma unroll
            for (int v = 0; v < 8; ++v) {
                float g = accG[m][n][v];
                float u = accU[m][n][v];
                float hv = (g / (1.0f + __expf(-g))) * u;
                int row = mBlk + wm * 32 + m * 16 + v + mo;
                int col = nBlk + wn * 64 + n * 16 + nn;
                hbuf[(size_t)row * IDIM + col] = f2bf(hv);
            }
}

// ---------------------------------------------------------------- GEMM 2
// out[T,H] += scale(t) * (h @ Wd). scale(t) = cw[t,e] (routed) or fixed (shared).
__global__ __launch_bounds__(256) void moe_down(const unsigned short* __restrict__ hbuf, // [T,I]
                                                const float* __restrict__ Wd,            // [I,H]
                                                float* __restrict__ out,                 // [T,H]
                                                const float* __restrict__ cw,            // [T,E_RT] or null
                                                int eidx, float fixedScale) {
    __shared__ unsigned short sA[2][128 * LSTR];
    __shared__ unsigned short sB[2][128 * LSTR];

    const int tid  = threadIdx.x;
    const int lane = tid & 31;
    const int wave = tid >> 5;
    const int wm = wave & 3;
    const int wn = wave >> 2;
    const int mBlk = blockIdx.y * 128;
    const int nBlk = blockIdx.x * 128;
    const int NK = IDIM / 32;

    const v8f vz = {0.f, 0.f, 0.f, 0.f, 0.f, 0.f, 0.f, 0.f};
    v8f acc[2][4];
#pragma unroll
    for (int m = 0; m < 2; ++m)
#pragma unroll
        for (int n = 0; n < 4; ++n) acc[m][n] = vz;

    stage_x(tid, mBlk, 0, hbuf, IDIM, sA[0]);
    stage_w(tid, nBlk, 0, Wd, HDIM, sB[0]);

    for (int kt = 0; kt < NK; ++kt) {
        lds_copy_wait();
        __syncthreads();
        const int b = kt & 1;

        if (kt + 1 < NK) {
            int k1 = (kt + 1) * 32;
            stage_x(tid, mBlk, k1, hbuf, IDIM, sA[b ^ 1]);
            stage_w(tid, nBlk, k1, Wd, HDIM, sB[b ^ 1]);
            if (kt + 2 < NK)
                __builtin_prefetch(&Wd[(size_t)(kt + 2) * 32 * HDIM + nBlk + tid], 0, 1);
        }

        const unsigned short* sa = sA[b];
        FragBF a[2];
#pragma unroll
        for (int m = 0; m < 2; ++m) {
            int row = wm * 32 + m * 16 + (lane & 15);
            int kh = (lane >> 4) << 4;
            a[m].q[0] = *(const uint4*)(sa + row * LSTR + kh);
            a[m].q[1] = *(const uint4*)(sa + row * LSTR + kh + 8);
        }
        const unsigned short* sb = sB[b];
#pragma unroll
        for (int n = 0; n < 4; ++n) {
            int colb = wn * 64 + n * 16 + (lane & 15);
            int kh = (lane >> 4) << 4;
            FragBF bb;
            const uint4* pb = (const uint4*)(sb + colb * LSTR + kh);
            bb.q[0] = pb[0]; bb.q[1] = pb[1];
#pragma unroll
            for (int m = 0; m < 2; ++m)
                acc[m][n] = __builtin_amdgcn_wmma_f32_16x16x32_bf16(
                    false, a[m].bf, false, bb.bf, (short)0, acc[m][n], false, false);
        }
    }

    const int mo = (lane >> 4) << 3;
    const int nn = lane & 15;
#pragma unroll
    for (int m = 0; m < 2; ++m)
#pragma unroll
        for (int n = 0; n < 4; ++n)
#pragma unroll
            for (int v = 0; v < 8; ++v) {
                int row = mBlk + wm * 32 + m * 16 + v + mo;
                int col = nBlk + wn * 64 + n * 16 + nn;
                float s = cw ? cw[row * NE_RT + eidx] : fixedScale;
                size_t oi = (size_t)row * HDIM + col;
                out[oi] += s * acc[m][n][v];  // unique tile per block; kernels stream-ordered
            }
}

// ---------------------------------------------------------------- launch
extern "C" void kernel_launch(void* const* d_in, const int* in_sizes, int n_in,
                              void* d_out, int out_size, void* d_ws, size_t ws_size,
                              hipStream_t stream) {
    const float* x    = (const float*)d_in[0];
    const float* Wg_s = (const float*)d_in[1];
    const float* Wu_s = (const float*)d_in[2];
    const float* Wd_s = (const float*)d_in[3];
    const float* Wg_r = (const float*)d_in[4];
    const float* Wu_r = (const float*)d_in[5];
    const float* Wd_r = (const float*)d_in[6];
    const float* Wr   = (const float*)d_in[7];
    const float* rb   = (const float*)d_in[8];
    float* out = (float*)d_out;

    // workspace layout (~23.2 MB): x_bf16 | cw | h_bf16
    char* ws = (char*)d_ws;
    size_t off = 0;
    unsigned short* xbf = (unsigned short*)(ws + off);
    off += (size_t)TTOK * HDIM * sizeof(unsigned short);
    off = (off + 255) & ~(size_t)255;
    float* cw = (float*)(ws + off);
    off += (size_t)TTOK * NE_RT * sizeof(float);
    off = (off + 255) & ~(size_t)255;
    unsigned short* hbuf = (unsigned short*)(ws + off);

    const int nTH = TTOK * HDIM;
    moe_zero_f32<<<(nTH + 255) / 256, 256, 0, stream>>>(out, nTH);
    moe_cvt_bf16<<<(nTH / 4 + 255) / 256, 256, 0, stream>>>(x, (unsigned int*)xbf, nTH);
    moe_router<<<TTOK / 8, 256, 0, stream>>>(x, Wr, rb, cw);

    const dim3 g1(IDIM / 128, TTOK / 128);  // 32 x 16
    const dim3 g2(HDIM / 128, TTOK / 128);  // 12 x 16
    const size_t wSz = (size_t)HDIM * IDIM;

    // shared experts: each contributes 1/E_SH
    for (int e = 0; e < NE_SH; ++e) {
        moe_gateup<<<g1, 256, 0, stream>>>(xbf, Wg_s + (size_t)e * wSz,
                                           Wu_s + (size_t)e * wSz, hbuf);
        moe_down<<<g2, 256, 0, stream>>>(hbuf, Wd_s + (size_t)e * wSz, out,
                                         nullptr, 0, 1.0f / NE_SH);
    }
    // routed experts: dense compute, combine-weight scaling (0 for unselected)
    for (int e = 0; e < NE_RT; ++e) {
        moe_gateup<<<g1, 256, 0, stream>>>(xbf, Wg_r + (size_t)e * wSz,
                                           Wu_r + (size_t)e * wSz, hbuf);
        moe_down<<<g2, 256, 0, stream>>>(hbuf, Wd_r + (size_t)e * wSz, out,
                                         cw, e, 0.0f);
    }
}
